// LoRALinear_76613626626548
// MI455X (gfx1250) — matrix operand order repaired and checked by reading, not compiled, not run on hardware
//
#include <hip/hip_runtime.h>
#include <stdint.h>

// ---------------------------------------------------------------------------
// LoRA linear for MI455X (gfx1250, wave32, WMMA).
// out = x @ W^T + bias + scale * ((x @ A_adapter^T, masked to rank) @ B_adapter)
// fp32 inputs -> exact bf16 hi + bf16 lo split (truncation; residual exact in
// f32), 3x v_wmma_f32_16x16x32_bf16 per K=32 step with f32 accumulation
// (bf16x3 ~ fp32 accuracy). Register double-buffered K loop, fused LoRA-B +
// bias epilogue.
// ---------------------------------------------------------------------------

typedef __attribute__((ext_vector_type(16))) __bf16 v16bf;
typedef __attribute__((ext_vector_type(8)))  float  v8f;

#define T_TOKENS   8192
#define DIM_IN     4096
#define DIM_OUT    4096
#define N_SEQS     4
#define MAX_RANK   64

#define BK   32     // K tile (fp32 elements) == WMMA K
#define LDA  40     // padded LDS stride in bf16 halves (80B rows, 16B aligned)

// ---- helpers --------------------------------------------------------------

struct U2  { unsigned int x, y; } __attribute__((aligned(8)));
struct U4  { unsigned int x, y, z, w; } __attribute__((aligned(16)));
struct Q32 { U4 a, b; };

// Exact split: hi = truncate-to-bf16(x) (top 16 bits), resid = x - hi is
// exact in f32; lo = truncate-to-bf16(resid). Two packed u32 per 4 values,
// one 8B LDS store each for hi and lo.
__device__ __forceinline__ void cvt_store4(float4 v, unsigned short* hi,
                                           unsigned short* lo) {
  unsigned u0 = __float_as_uint(v.x), u1 = __float_as_uint(v.y);
  unsigned u2 = __float_as_uint(v.z), u3 = __float_as_uint(v.w);
  float r0 = v.x - __uint_as_float(u0 & 0xFFFF0000u);
  float r1 = v.y - __uint_as_float(u1 & 0xFFFF0000u);
  float r2 = v.z - __uint_as_float(u2 & 0xFFFF0000u);
  float r3 = v.w - __uint_as_float(u3 & 0xFFFF0000u);
  U2 h, l;
  h.x = (u0 >> 16) | (u1 & 0xFFFF0000u);
  h.y = (u2 >> 16) | (u3 & 0xFFFF0000u);
  l.x = (__float_as_uint(r0) >> 16) | (__float_as_uint(r1) & 0xFFFF0000u);
  l.y = (__float_as_uint(r2) >> 16) | (__float_as_uint(r3) & 0xFFFF0000u);
  *(U2*)hi = h;   // ds_store_b64
  *(U2*)lo = l;   // ds_store_b64
}

// two aligned 16B LDS reads -> one 16x32 / 32x16 bf16 WMMA fragment
__device__ __forceinline__ v16bf ldfrag(const unsigned short* p0,
                                        const unsigned short* p1) {
  Q32 q;
  q.a = *(const U4*)p0;   // ds_load_b128
  q.b = *(const U4*)p1;   // ds_load_b128
  return __builtin_bit_cast(v16bf, q);
}

__device__ __forceinline__ v8f wmma_bf(v16bf a, v16bf b, v8f c) {
  // (neg_a, A, neg_b, B, c_mod, C, reuse_a, reuse_b)
  return __builtin_amdgcn_wmma_f32_16x16x32_bf16(false, a, false, b,
                                                 (short)0, c, false, false);
}

__device__ __forceinline__ int seq_of(const int* __restrict__ start, int t) {
  int s = 0;
#pragma unroll
  for (int i = 1; i < N_SEQS; ++i)
    if (start[i] <= t) s = i;
  return s;
}

// ---------------------------------------------------------------------------
// Kernel 1: xa[t, r] = x[t, :] . a_cache[page(adapter(t), r), :]   (0 if r>=rank)
// Block: 32 tokens x 64 ranks, 256 threads = 8 waves (2 M x 4 R tiles).
// ---------------------------------------------------------------------------
__global__ __launch_bounds__(256)
void lora_xa_kernel(const float* __restrict__ x,
                    const float* __restrict__ a_cache,
                    const int*   __restrict__ b_start_loc,
                    const int*   __restrict__ b_adapter_ids,
                    const int*   __restrict__ rank_page_table,
                    const int*   __restrict__ ranks,
                    float*       __restrict__ xa) {
  __shared__ unsigned short xs_hi[32 * LDA], xs_lo[32 * LDA];
  __shared__ unsigned short as_hi[64 * LDA], as_lo[64 * LDA];

  const int tid   = threadIdx.x;
  const int lane  = tid & 31;
  const int wave  = tid >> 5;
  const int lr    = lane >> 4;       // half-wave index
  const int lc    = lane & 15;
  const int waveM = wave >> 2;       // 0..1 : 16-token tile
  const int waveR = wave & 3;        // 0..3 : 16-rank tile
  const int bm    = blockIdx.x * 32; // 2048 % 32 == 0 -> one adapter per block

  const int seq     = seq_of(b_start_loc, bm);
  const int adapter = b_adapter_ids[seq];
  const int rank    = ranks[adapter];

  v8f acc = {};

  const int srow = tid >> 3;          // 0..31
  const int scol = (tid & 7) * 4;     // 0..28

  // paged A row addresses for this thread's two staging rows
  const int   page0 = rank_page_table[adapter * MAX_RANK + srow];
  const int   page1 = rank_page_table[adapter * MAX_RANK + 32 + srow];
  const float* xb = &x[(size_t)(bm + srow) * DIM_IN + scol];
  const float* ab0 = &a_cache[(size_t)page0 * DIM_IN + scol];
  const float* ab1 = &a_cache[(size_t)page1 * DIM_IN + scol];

  float4 rx, ra0, ra1;
  auto load_tile = [&](int k0) {
    rx  = *(const float4*)(xb  + k0);
    ra0 = *(const float4*)(ab0 + k0);
    ra1 = *(const float4*)(ab1 + k0);
  };
  auto store_tile = [&]() {
    cvt_store4(rx,  &xs_hi[srow * LDA + scol],        &xs_lo[srow * LDA + scol]);
    cvt_store4(ra0, &as_hi[srow * LDA + scol],        &as_lo[srow * LDA + scol]);
    cvt_store4(ra1, &as_hi[(32 + srow) * LDA + scol], &as_lo[(32 + srow) * LDA + scol]);
  };

  load_tile(0);
  store_tile();
  __syncthreads();

  for (int k0 = 0; k0 < DIM_IN; k0 += BK) {
    const bool more = (k0 + BK) < DIM_IN;
    if (more) load_tile(k0 + BK);   // prefetch next tile into registers

    {
      // A fragment: 16x32 bf16, lane lc = row, lr selects K octets
      const int arow = waveM * 16 + lc;
      const unsigned short* pah = &xs_hi[arow * LDA + lr * 8];
      const unsigned short* pal = &xs_lo[arow * LDA + lr * 8];
      v16bf ah = ldfrag(pah, pah + 16);
      v16bf al = ldfrag(pal, pal + 16);
      // B fragment: 32x16 bf16, lane lc = column (rank), lr selects K octets
      const int bcol = waveR * 16 + lc;
      const unsigned short* pbh = &as_hi[bcol * LDA + lr * 8];
      const unsigned short* pbl = &as_lo[bcol * LDA + lr * 8];
      v16bf bh = ldfrag(pbh, pbh + 16);
      v16bf bl = ldfrag(pbl, pbl + 16);

      acc = wmma_bf(ah, bh, acc);   // hi*hi
      acc = wmma_bf(al, bh, acc);   // lo*hi
      acc = wmma_bf(ah, bl, acc);   // hi*lo
    }
    __syncthreads();                // all waves done reading this tile
    if (more) {
      store_tile();
      __syncthreads();              // stores visible before next compute
    }
  }

  // D layout: VGPR v, lane -> (M = v + 8*lr, N = lc)
#pragma unroll
  for (int v = 0; v < 8; ++v) {
    int t = bm + waveM * 16 + lr * 8 + v;
    int r = waveR * 16 + lc;
    xa[(size_t)t * MAX_RANK + r] = (r < rank) ? acc[v] : 0.0f;
  }
}

// ---------------------------------------------------------------------------
// Kernel 2: out = x @ W^T + bias + scale * (xa @ B_adapter)
// Block: 128x128 output tile, 256 threads = 8 waves (4 M x 2 N slabs),
// each wave owns 2x4 = 8 WMMA 16x16 accumulators; 24 WMMA per K=32 step.
// ---------------------------------------------------------------------------
#define BM2 128
#define BN2 128

__global__ __launch_bounds__(256)
void lora_main_kernel(const float* __restrict__ x,
                      const float* __restrict__ weight,
                      const float* __restrict__ bias,
                      const float* __restrict__ b_cache,
                      const int*   __restrict__ b_start_loc,
                      const int*   __restrict__ b_adapter_ids,
                      const float* __restrict__ b_scaling,
                      const int*   __restrict__ rank_page_table,
                      const float* __restrict__ xa,
                      float*       __restrict__ out) {
  __shared__ union {
    struct {  // main-loop staging: 40 KB
      unsigned short xs_hi[BM2 * LDA], xs_lo[BM2 * LDA];
      unsigned short ws_hi[BN2 * LDA], ws_lo[BN2 * LDA];
    } mm;
    struct {  // epilogue staging: 64 KB
      float xa_s[BM2 * MAX_RANK];
      float bs[MAX_RANK * BN2];     // pre-scaled B rows
    } ep;
  } sh;

  const int tid   = threadIdx.x;
  const int lane  = tid & 31;
  const int wave  = tid >> 5;
  const int lr    = lane >> 4;
  const int lc    = lane & 15;
  const int waveM = wave >> 1;       // 0..3 : 32-row slab
  const int waveN = wave & 1;        // 0..1 : 64-col slab
  const int bm    = blockIdx.y * BM2;
  const int bn    = blockIdx.x * BN2;

  v8f acc[2][4];
#pragma unroll
  for (int mt = 0; mt < 2; ++mt)
#pragma unroll
    for (int nt = 0; nt < 4; ++nt)
      acc[mt][nt] = (v8f){};

  const int srow = tid >> 3;          // 0..31
  const int scol = (tid & 7) * 4;     // 0..28

  const float* xb = &x[(size_t)(bm + srow) * DIM_IN + scol];
  const float* wb = &weight[(size_t)(bn + srow) * DIM_IN + scol];

  float4 rx[4], rw[4];
  auto load_tile = [&](int k0) {      // 8x global_load_b128, issued early
#pragma unroll
    for (int p = 0; p < 4; ++p) {
      rx[p] = *(const float4*)(xb + (size_t)(p * 32) * DIM_IN + k0);
      rw[p] = *(const float4*)(wb + (size_t)(p * 32) * DIM_IN + k0);
    }
  };
  auto store_tile = [&]() {           // cvt + packed ds_store_b64
#pragma unroll
    for (int p = 0; p < 4; ++p) {
      const int row = p * 32 + srow;
      cvt_store4(rx[p], &sh.mm.xs_hi[row * LDA + scol], &sh.mm.xs_lo[row * LDA + scol]);
      cvt_store4(rw[p], &sh.mm.ws_hi[row * LDA + scol], &sh.mm.ws_lo[row * LDA + scol]);
    }
  };

  load_tile(0);
  store_tile();
  __syncthreads();

  for (int k0 = 0; k0 < DIM_IN; k0 += BK) {
    const bool more = (k0 + BK) < DIM_IN;
    if (more) load_tile(k0 + BK);    // hide global latency behind WMMAs

    v16bf ah[2], al[2];
#pragma unroll
    for (int mt = 0; mt < 2; ++mt) {
      const int arow = waveM * 32 + mt * 16 + lc;
      const unsigned short* pah = &sh.mm.xs_hi[arow * LDA + lr * 8];
      const unsigned short* pal = &sh.mm.xs_lo[arow * LDA + lr * 8];
      ah[mt] = ldfrag(pah, pah + 16);
      al[mt] = ldfrag(pal, pal + 16);
    }
#pragma unroll
    for (int nt = 0; nt < 4; ++nt) {
      const int bcol = waveN * 64 + nt * 16 + lc;
      const unsigned short* pbh = &sh.mm.ws_hi[bcol * LDA + lr * 8];
      const unsigned short* pbl = &sh.mm.ws_lo[bcol * LDA + lr * 8];
      v16bf bh = ldfrag(pbh, pbh + 16);
      v16bf bl = ldfrag(pbl, pbl + 16);
#pragma unroll
      for (int mt = 0; mt < 2; ++mt) {
        acc[mt][nt] = wmma_bf(ah[mt], bh, acc[mt][nt]);   // hi*hi
        acc[mt][nt] = wmma_bf(al[mt], bh, acc[mt][nt]);   // lo*hi
        acc[mt][nt] = wmma_bf(ah[mt], bl, acc[mt][nt]);   // hi*lo
      }
    }
    __syncthreads();                 // all waves done reading this tile
    if (more) {
      store_tile();
      __syncthreads();               // stores visible before next compute
    }
  }

  // ---- fused LoRA-B + bias epilogue --------------------------------------
  const int   seq     = seq_of(b_start_loc, bm);   // 2048 % 128 == 0
  const int   adapter = b_adapter_ids[seq];
  const float scale   = b_scaling[seq];

  // stage xa block [128 x 64] (already zero beyond rank)
  for (int i = tid; i < BM2 * (MAX_RANK / 4); i += 256) {
    int row = i >> 4, c4 = (i & 15) << 2;
    *(float4*)&sh.ep.xa_s[row * MAX_RANK + c4] =
        *(const float4*)&xa[(size_t)(bm + row) * MAX_RANK + c4];
  }
  // stage scale * B rows [64 x 128]
  for (int i = tid; i < MAX_RANK * (BN2 / 4); i += 256) {
    int r = i >> 5, c4 = (i & 31) << 2;
    int page = rank_page_table[adapter * MAX_RANK + r];
    float4 v = *(const float4*)&b_cache[(size_t)page * DIM_OUT + bn + c4];
    v.x *= scale; v.y *= scale; v.z *= scale; v.w *= scale;
    *(float4*)&sh.ep.bs[r * BN2 + c4] = v;
  }
  __syncthreads();

  // rank-64 outer-product accumulation into the WMMA accumulators
  for (int r = 0; r < MAX_RANK; ++r) {
    float bv[4];
#pragma unroll
    for (int nt = 0; nt < 4; ++nt)
      bv[nt] = sh.ep.bs[r * BN2 + waveN * 64 + nt * 16 + lc];
#pragma unroll
    for (int mt = 0; mt < 2; ++mt) {
#pragma unroll
      for (int v = 0; v < 8; ++v) {
        float xv = sh.ep.xa_s[(waveM * 32 + mt * 16 + lr * 8 + v) * MAX_RANK + r];
#pragma unroll
        for (int nt = 0; nt < 4; ++nt)
          acc[mt][nt][v] += xv * bv[nt];
      }
    }
  }

  // store with bias
  float bb[4];
#pragma unroll
  for (int nt = 0; nt < 4; ++nt)
    bb[nt] = bias[bn + waveN * 64 + nt * 16 + lc];

#pragma unroll
  for (int mt = 0; mt < 2; ++mt)
#pragma unroll
    for (int v = 0; v < 8; ++v) {
      const size_t row = (size_t)(bm + waveM * 32 + mt * 16 + lr * 8 + v);
#pragma unroll
      for (int nt = 0; nt < 4; ++nt) {
        const int col = bn + waveN * 64 + nt * 16 + lc;
        out[row * DIM_OUT + col] = acc[mt][nt][v] + bb[nt];
      }
    }
}

// ---------------------------------------------------------------------------

extern "C" void kernel_launch(void* const* d_in, const int* in_sizes, int n_in,
                              void* d_out, int out_size, void* d_ws, size_t ws_size,
                              hipStream_t stream) {
  const float* x               = (const float*)d_in[0];
  const float* weight          = (const float*)d_in[1];
  const float* bias            = (const float*)d_in[2];
  const float* a_cache         = (const float*)d_in[3];
  const float* b_cache         = (const float*)d_in[4];
  const int*   b_start_loc     = (const int*)d_in[5];
  const int*   b_adapter_ids   = (const int*)d_in[6];
  const float* b_scaling       = (const float*)d_in[7];
  const int*   rank_page_table = (const int*)d_in[8];
  const int*   ranks           = (const int*)d_in[9];
  float*       out             = (float*)d_out;
  float*       xa_ws           = (float*)d_ws;   // T_TOKENS * MAX_RANK floats (2 MB)

  lora_xa_kernel<<<dim3(T_TOKENS / 32), dim3(256), 0, stream>>>(
      x, a_cache, b_start_loc, b_adapter_ids, rank_page_table, ranks, xa_ws);

  lora_main_kernel<<<dim3(DIM_OUT / BN2, T_TOKENS / BM2), dim3(256), 0, stream>>>(
      x, weight, bias, b_cache, b_start_loc, b_adapter_ids, b_scaling,
      rank_page_table, xa_ws, out);
}
